// FNOGNO_58463094833807
// MI455X (gfx1250) — matrix operand order repaired and checked by reading, not compiled, not run on hardware
//
#include <hip/hip_runtime.h>
#include <math.h>

// ---------------------------------------------------------------------------
// FNO-GNO pipeline for gfx1250 (MI455X). All dense GEMMs use
// v_wmma_f32_16x16x32_bf16 (fp32 accumulate), double-buffered LDS staging,
// packed v_cvt_pk_bf16_f32 conversion. Spectral conv = truncated DFTs with
// LDS twiddle tables.
// ---------------------------------------------------------------------------

typedef __attribute__((ext_vector_type(16))) __bf16 v16bf;
typedef __attribute__((ext_vector_type(2)))  __bf16 v2bf;
typedef __attribute__((ext_vector_type(8)))  float  v8f;
typedef __attribute__((ext_vector_type(2)))  float  v2f;

#define GSZ    32
#define NIN    32768        // 32^3
#define NOUTP  16384
#define EDGES  262144
#define ECHUNK 16384
#define PI2F   6.28318530717958647692f

// packed fp32x2 -> bf16x2 (RNE) -> one dword; lowers to v_cvt_pk_bf16_f32
__device__ __forceinline__ unsigned pack_bf16(float lo, float hi) {
  v2f t; t.x = lo; t.y = hi;
  union { v2bf v; unsigned u; } p;
  p.v = __builtin_convertvector(t, v2bf);
  return p.u;
}
__device__ __forceinline__ unsigned short f2bf(float f) {
  union { __bf16 b; unsigned short u; } p;
  p.b = (__bf16)f;
  return p.u;
}
__device__ __forceinline__ float gelu_exact(float x) {
  return 0.5f * x * (1.0f + erff(x * 0.70710678118654752440f));
}

// ------------------------- small utility kernels ---------------------------

__global__ void k_zero(float* __restrict__ p, int n) {
  int t = blockIdx.x * 256 + threadIdx.x;
  if (t < n) p[t] = 0.0f;
}

// Bt[n*Kpad + k] = bf16(W[n*K + k]), zero-padded k in [K,Kpad)
__global__ void k_wT(const float* __restrict__ W, unsigned short* __restrict__ Bt,
                     int N, int K, int Kpad) {
  int t = blockIdx.x * 256 + threadIdx.x;
  if (t < N * Kpad) {
    int n = t / Kpad, k = t % Kpad;
    Bt[t] = (k < K) ? f2bf(W[(size_t)n * K + k]) : (unsigned short)0;
  }
}

// A6[n, 0:3]=f[n], [3:6]=in_p[n], [6:32]=0   (row stride 32)
__global__ void k_build_a6(const float* __restrict__ f, const float* __restrict__ in_p,
                           float* __restrict__ A6) {
  int t = blockIdx.x * 256 + threadIdx.x;
  if (t < NIN) {
    float4* row = (float4*)(A6 + (size_t)t * 32);
    float4 a = {f[t * 3 + 0], f[t * 3 + 1], f[t * 3 + 2], in_p[t * 3 + 0]};
    float4 b = {in_p[t * 3 + 1], in_p[t * 3 + 2], 0.0f, 0.0f};
    float4 z = {0.0f, 0.0f, 0.0f, 0.0f};
    row[0] = a; row[1] = b;
    #pragma unroll
    for (int j = 2; j < 8; ++j) row[j] = z;
  }
}

// feats[e, 0:3]=in_p[nbi[e]], [3:6]=out_p[seg[e]], [6:32]=0  (row stride 32)
__global__ void k_build_feats(const float* __restrict__ in_p, const float* __restrict__ out_p,
                              const int* __restrict__ nbi, const int* __restrict__ seg,
                              float* __restrict__ feats) {
  int t = blockIdx.x * 256 + threadIdx.x;
  if (t < EDGES) {
    int s = nbi[t], d = seg[t];
    float4* row = (float4*)(feats + (size_t)t * 32);
    float4 a = {in_p[(size_t)s * 3 + 0], in_p[(size_t)s * 3 + 1],
                in_p[(size_t)s * 3 + 2], out_p[(size_t)d * 3 + 0]};
    float4 b = {out_p[(size_t)d * 3 + 1], out_p[(size_t)d * 3 + 2], 0.0f, 0.0f};
    float4 z = {0.0f, 0.0f, 0.0f, 0.0f};
    row[0] = a; row[1] = b;
    #pragma unroll
    for (int j = 2; j < 8; ++j) row[j] = z;
  }
}

// ----------------------------- WMMA GEMM -----------------------------------
// C[M,N] = act( A[M,K](fp32, row stride K) * Bt[N,K](bf16) + bias + addend )
// Preconditions: K%32==0, M%128==0, N%64==0.
// 256 threads = 8 waves; block tile 128x64; each wave: 4 x (16x16) D tiles.
// Double-buffered LDS staging: global loads for tile t+1 issue before the
// WMMAs of tile t; LDS stores go to the other buffer after them.
#define LDSP 20   // padded row stride (dwords): 80B, 16B-aligned, bank-spread
__launch_bounds__(256)
__global__ void k_gemm(const float* __restrict__ A, const unsigned short* __restrict__ Bt,
                       const float* __restrict__ bias, const float* __restrict__ addend,
                       float* __restrict__ Cmat, int M, int N, int K, int act) {
  __shared__ unsigned int sA[2][128 * LDSP];  // per row: 16 dwords = 32 bf16 (k)
  __shared__ unsigned int sB[2][64 * LDSP];   // per col: 16 dwords = 32 bf16 (k)
  const int tid  = threadIdx.x;
  const int lane = tid & 31;
  const int w    = tid >> 5, wr = w >> 2, wc = w & 3;
  const int bm0  = blockIdx.y * 128;
  const int bn0  = blockIdx.x * 64;
  const int lo   = lane & 15, hi = lane >> 4;
  const int cN   = tid >> 2, gB = tid & 3;    // B staging: col, 8-bf16 group

  v8f acc[4];
  #pragma unroll
  for (int s = 0; s < 4; ++s)
    #pragma unroll
    for (int r = 0; r < 8; ++r) acc[s][r] = 0.0f;

  float4 aregs[4];
  uint4  breg;

  // gather staging tile (global -> registers), all loads issued back-to-back
  auto tile_load = [&](int k0) {
    #pragma unroll
    for (int i = 0; i < 4; ++i) {
      int idx = tid + 256 * i;          // 1024 float4 groups
      int r = idx >> 3, g = idx & 7;    // row, group of 4 floats
      aregs[i] = *(const float4*)(A + (size_t)(bm0 + r) * K + k0 + g * 4);
    }
    breg = *(const uint4*)(Bt + (size_t)(bn0 + cN) * K + k0 + gB * 8);
  };
  // registers -> LDS buffer (packed bf16 convert for A)
  auto tile_store = [&](int buf) {
    #pragma unroll
    for (int i = 0; i < 4; ++i) {
      int idx = tid + 256 * i;
      int r = idx >> 3, g = idx & 7;
      sA[buf][r * LDSP + g * 2 + 0] = pack_bf16(aregs[i].x, aregs[i].y);
      sA[buf][r * LDSP + g * 2 + 1] = pack_bf16(aregs[i].z, aregs[i].w);
    }
    sB[buf][cN * LDSP + gB * 4 + 0] = breg.x;
    sB[buf][cN * LDSP + gB * 4 + 1] = breg.y;
    sB[buf][cN * LDSP + gB * 4 + 2] = breg.z;
    sB[buf][cN * LDSP + gB * 4 + 3] = breg.w;
  };

  const int nk = K >> 5;
  tile_load(0);
  tile_store(0);
  __syncthreads();

  union frag { v16bf v; unsigned u[8]; };

  for (int t = 0; t < nk; ++t) {
    const int cur = t & 1;
    if (t + 1 < nk) tile_load((t + 1) << 5);   // overlap with WMMAs below

    // B fragment: col = wc*16+lo; k = hi*8 + 2q (q<4), 16 + hi*8 + 2(q-4) (q>=4)
    frag bfrag;
    #pragma unroll
    for (int q = 0; q < 8; ++q)
      bfrag.u[q] = sB[cur][(wc * 16 + lo) * LDSP + (q >> 2) * 8 + hi * 4 + (q & 3)];

    // load ALL A fragments first so the ds loads overlap, then issue WMMAs
    frag afrag[4];
    #pragma unroll
    for (int s = 0; s < 4; ++s) {
      int m = wr * 64 + s * 16 + lo;
      #pragma unroll
      for (int q = 0; q < 4; ++q) {
        afrag[s].u[q]     = sA[cur][m * LDSP + hi * 4 + q];     // k = hi*8 + 2q
        afrag[s].u[4 + q] = sA[cur][m * LDSP + 8 + hi * 4 + q]; // k = 16 + hi*8 + 2q
      }
    }
    #pragma unroll
    for (int s = 0; s < 4; ++s)
      acc[s] = __builtin_amdgcn_wmma_f32_16x16x32_bf16(
          false, afrag[s].v, false, bfrag.v, (short)0, acc[s], false, false);

    if (t + 1 < nk) tile_store(1 - cur);
    __syncthreads();
  }

  #pragma unroll
  for (int s = 0; s < 4; ++s) {
    int col = bn0 + wc * 16 + lo;
    #pragma unroll
    for (int r = 0; r < 8; ++r) {
      int row = bm0 + wr * 64 + s * 16 + r + hi * 8;
      float x = acc[s][r];
      if (bias)   x += bias[col];
      if (addend) x += addend[(size_t)row * N + col];
      if (act)    x = gelu_exact(x);
      Cmat[(size_t)row * N + col] = x;
    }
  }
}

// ------------------- LDS twiddle table helper (angles k*2pi/32) ------------
#define MAKE_TWIDDLE(tc, ts)                                        \
  __shared__ float tc[32]; __shared__ float ts[32];                 \
  if (threadIdx.x < 32) {                                           \
    float th = PI2F * (float)threadIdx.x * (1.0f / 32.0f);          \
    tc[threadIdx.x] = cosf(th); ts[threadIdx.x] = sinf(th);         \
  }                                                                 \
  __syncthreads();

// --------------------- truncated forward DFT (rfftn) -----------------------
// x layout: (gx*32+gy)*32+gz spatial, C=64 innermost. norm='forward' => 1/32768.

__global__ void k_dft_z(const float* __restrict__ x, float2* __restrict__ T1) {
  MAKE_TWIDDLE(tc, ts)
  int blk = blockIdx.x;                 // (gx*32+gy) in [0,1024)
  int c = threadIdx.x & 63, kzg = threadIdx.x >> 6;
  const float norm = 1.0f / 32768.0f;
  for (int j = 0; j < 2; ++j) {
    int kz = kzg * 2 + j;
    float ar = 0.f, ai = 0.f;
    for (int z = 0; z < GSZ; ++z) {
      float v = x[((size_t)blk * 32 + z) * 64 + c];
      int a = (kz * z) & 31;
      ar += v * tc[a];
      ai -= v * ts[a];
    }
    T1[((size_t)blk * 8 + kz) * 64 + c] = make_float2(ar * norm, ai * norm);
  }
}

__global__ void k_dft_y(const float2* __restrict__ T1, float2* __restrict__ T2) {
  MAKE_TWIDDLE(tc, ts)
  int blk = blockIdx.x;                 // gx*16+kyi in [0,512)
  int xg = blk >> 4, kyi = blk & 15;
  int ky = (kyi < 8) ? kyi : kyi + 16;  // 0..7 or 24..31
  int c = threadIdx.x & 63, kzg = threadIdx.x >> 6;
  for (int j = 0; j < 2; ++j) {
    int kz = kzg * 2 + j;
    float ar = 0.f, ai = 0.f;
    for (int y = 0; y < GSZ; ++y) {
      float2 t = T1[(((size_t)xg * 32 + y) * 8 + kz) * 64 + c];
      int a = (ky * y) & 31;
      float cs = tc[a], sn = ts[a];
      ar += t.x * cs + t.y * sn;        // * e^{-i th}
      ai += t.y * cs - t.x * sn;
    }
    T2[(((size_t)xg * 16 + kyi) * 8 + kz) * 64 + c] = make_float2(ar, ai);
  }
}

__global__ void k_dft_x(const float2* __restrict__ T2, float2* __restrict__ Xf) {
  MAKE_TWIDDLE(tc, ts)
  int blk = blockIdx.x;                 // kxi*16+kyi in [0,256)
  int kxi = blk >> 4, kyi = blk & 15;
  int kx = (kxi < 8) ? kxi : kxi + 16;
  int c = threadIdx.x & 63, kzg = threadIdx.x >> 6;
  for (int j = 0; j < 2; ++j) {
    int kz = kzg * 2 + j;
    float ar = 0.f, ai = 0.f;
    for (int xg = 0; xg < GSZ; ++xg) {
      float2 t = T2[(((size_t)xg * 16 + kyi) * 8 + kz) * 64 + c];
      int a = (kx * xg) & 31;
      float cs = tc[a], sn = ts[a];
      ar += t.x * cs + t.y * sn;
      ai += t.y * cs - t.x * sn;
    }
    Xf[(((size_t)kxi * 16 + kyi) * 8 + kz) * 64 + c] = make_float2(ar, ai);
  }
}

// ---------------- per-mode 64x64 complex matvec (corner weights) -----------
__global__ void k_modemul(const float2* __restrict__ Xf, const float* __restrict__ wr,
                          const float* __restrict__ wi, float2* __restrict__ Yf) {
  __shared__ float2 sx[64];
  int m = blockIdx.x;                   // (kxi*16+kyi)*8+kz in [0,2048)
  int kxi = m >> 7, kyi = (m >> 3) & 15, kz = m & 7;
  int k = (kxi >= 8 ? 2 : 0) + (kyi >= 8 ? 1 : 0);   // corner index
  int moff = (kxi & 7) * 64 + (kyi & 7) * 8 + kz;    // mx*64+my*8+mz
  int o = threadIdx.x;                  // 64 threads
  sx[o] = Xf[(size_t)m * 64 + o];
  __syncthreads();
  float yr = 0.f, yi = 0.f;
  size_t base = ((size_t)k * 64) * 64 * 512 + (size_t)o * 512 + moff;
  for (int i = 0; i < 64; ++i) {
    float a = wr[base + (size_t)i * (64 * 512)];
    float b = wi[base + (size_t)i * (64 * 512)];
    yr += a * sx[i].x - b * sx[i].y;
    yi += a * sx[i].y + b * sx[i].x;
  }
  Yf[(size_t)m * 64 + o] = make_float2(yr, yi);
}

// ----------------------- truncated inverse DFT -----------------------------

__global__ void k_idft_x(const float2* __restrict__ Yf, float2* __restrict__ U1) {
  MAKE_TWIDDLE(tc, ts)
  int blk = blockIdx.x;                 // xg*16+kyi in [0,512)
  int xg = blk >> 4, kyi = blk & 15;
  int c = threadIdx.x & 63, kzg = threadIdx.x >> 6;
  for (int j = 0; j < 2; ++j) {
    int kz = kzg * 2 + j;
    float ar = 0.f, ai = 0.f;
    #pragma unroll
    for (int kxi = 0; kxi < 16; ++kxi) {
      int kx = (kxi < 8) ? kxi : kxi + 16;
      float2 t = Yf[(((size_t)kxi * 16 + kyi) * 8 + kz) * 64 + c];
      int a = (kx * xg) & 31;
      float cs = tc[a], sn = ts[a];
      ar += t.x * cs - t.y * sn;        // * e^{+i th}
      ai += t.x * sn + t.y * cs;
    }
    U1[(((size_t)xg * 16 + kyi) * 8 + kz) * 64 + c] = make_float2(ar, ai);
  }
}

__global__ void k_idft_y(const float2* __restrict__ U1, float2* __restrict__ U2) {
  MAKE_TWIDDLE(tc, ts)
  int blk = blockIdx.x;                 // xg*32+y in [0,1024)
  int xg = blk >> 5, y = blk & 31;
  int c = threadIdx.x & 63, kzg = threadIdx.x >> 6;
  for (int j = 0; j < 2; ++j) {
    int kz = kzg * 2 + j;
    float ar = 0.f, ai = 0.f;
    #pragma unroll
    for (int kyi = 0; kyi < 16; ++kyi) {
      int ky = (kyi < 8) ? kyi : kyi + 16;
      float2 t = U1[(((size_t)xg * 16 + kyi) * 8 + kz) * 64 + c];
      int a = (ky * y) & 31;
      float cs = tc[a], sn = ts[a];
      ar += t.x * cs - t.y * sn;
      ai += t.x * sn + t.y * cs;
    }
    U2[((size_t)blk * 8 + kz) * 64 + c] = make_float2(ar, ai);
  }
}

// c2r along z (half-spectrum, only kz=0..7 nonzero; Im of bin0 ignored)
__global__ void k_idft_z(const float2* __restrict__ U2, float* __restrict__ sres) {
  MAKE_TWIDDLE(tc, ts)
  int blk = blockIdx.x;                 // (gx*32+gy) in [0,1024)
  int c = threadIdx.x & 63, zg = threadIdx.x >> 6;
  float2 u[8];
  #pragma unroll
  for (int kz = 0; kz < 8; ++kz) u[kz] = U2[((size_t)blk * 8 + kz) * 64 + c];
  for (int t = 0; t < 8; ++t) {
    int z = zg * 8 + t;
    float val = u[0].x;
    #pragma unroll
    for (int kz = 1; kz < 8; ++kz) {
      int a = (kz * z) & 31;
      val += 2.0f * (u[kz].x * tc[a] - u[kz].y * ts[a]);
    }
    sres[((size_t)blk * 32 + z) * 64 + c] = val;
  }
}

// --------------------------- GNO scatter / reduce --------------------------

__global__ void k_counts(const int* __restrict__ seg, float* __restrict__ cnt) {
  int t = blockIdx.x * 256 + threadIdx.x;
  if (t < EDGES) atomicAdd(&cnt[seg[t]], 1.0f);
}

__global__ void k_scatter(const float* __restrict__ kern, const float* __restrict__ latent,
                          const int* __restrict__ nbi, const int* __restrict__ seg,
                          float* __restrict__ ssum, int e0) {
  int t = blockIdx.x * 256 + threadIdx.x;       // ECHUNK*64 threads
  int ei = t >> 6, c = t & 63;
  int e = e0 + ei;
  float v = kern[(size_t)ei * 64 + c] * latent[(size_t)nbi[e] * 64 + c];
  atomicAdd(&ssum[(size_t)seg[e] * 64 + c], v);
}

__global__ void k_mean(const float* __restrict__ ssum, const float* __restrict__ cnt,
                       float* __restrict__ outf) {
  int t = blockIdx.x * 256 + threadIdx.x;
  if (t < NOUTP * 64) outf[t] = ssum[t] / fmaxf(cnt[t >> 6], 1.0f);
}

// final: out[row] = sum_c P[row,c]*w2[c] + b2   (one wave per row)
__global__ void k_proj2(const float* __restrict__ P, const float* __restrict__ w2,
                        const float* __restrict__ b2, float* __restrict__ out) {
  int gid = blockIdx.x * 256 + threadIdx.x;
  int row = gid >> 5, lane = gid & 31;
  if (row >= NOUTP) return;
  float s = 0.0f;
  #pragma unroll
  for (int i = 0; i < 8; ++i) {
    int c = lane + 32 * i;
    s += P[(size_t)row * 256 + c] * w2[c];
  }
  #pragma unroll
  for (int m = 16; m > 0; m >>= 1) s += __shfl_xor(s, m, 32);
  if (lane == 0) out[row] = s + b2[0];
}

// ---------------------------------------------------------------------------

static inline int cdiv(int a, int b) { return (a + b - 1) / b; }

extern "C" void kernel_launch(void* const* d_in, const int* in_sizes, int n_in,
                              void* d_out, int out_size, void* d_ws, size_t ws_size,
                              hipStream_t stream) {
  const float* in_p    = (const float*)d_in[0];
  const float* out_p   = (const float*)d_in[1];
  const float* f       = (const float*)d_in[2];
  const int*   nbi     = (const int*)  d_in[3];
  const int*   seg     = (const int*)  d_in[4];
  const float* lift_w1 = (const float*)d_in[5];
  const float* lift_b1 = (const float*)d_in[6];
  const float* lift_w2 = (const float*)d_in[7];
  const float* lift_b2 = (const float*)d_in[8];
  const float* spec_wr = (const float*)d_in[9];
  const float* spec_wi = (const float*)d_in[10];
  const float* skip_w  = (const float*)d_in[11];
  const float* skip_b  = (const float*)d_in[12];
  const float* gno_w1  = (const float*)d_in[13];
  const float* gno_b1  = (const float*)d_in[14];
  const float* gno_w2  = (const float*)d_in[15];
  const float* gno_b2  = (const float*)d_in[16];
  const float* gno_w3  = (const float*)d_in[17];
  const float* gno_b3  = (const float*)d_in[18];
  const float* proj_w1 = (const float*)d_in[19];
  const float* proj_b1 = (const float*)d_in[20];
  const float* proj_w2 = (const float*)d_in[21];
  const float* proj_b2 = (const float*)d_in[22];
  float* out = (float*)d_out;

  char* ws = (char*)d_ws;
  size_t off = 0;
  auto alloc = [&](size_t bytes) -> size_t {
    size_t o = off; off = (off + bytes + 255) & ~(size_t)255; return o;
  };
  // persistent buffers
  float* xA    = (float*)(ws + alloc((size_t)NIN * 64 * 4));
  float* xB    = (float*)(ws + alloc((size_t)NIN * 64 * 4));
  float* A6    = (float*)(ws + alloc((size_t)NIN * 32 * 4));      // K padded to 32
  float* feats = (float*)(ws + alloc((size_t)EDGES * 32 * 4));    // K padded to 32
  float* ssum  = (float*)(ws + alloc((size_t)NOUTP * 64 * 4));
  float* cnt   = (float*)(ws + alloc((size_t)NOUTP * 4));
  float* outf  = (float*)(ws + alloc((size_t)NOUTP * 64 * 4));
  float* Pj    = (float*)(ws + alloc((size_t)NOUTP * 256 * 4));
  unsigned short* wl1 = (unsigned short*)(ws + alloc((size_t)256 * 32 * 2));
  unsigned short* wl2 = (unsigned short*)(ws + alloc((size_t)64 * 256 * 2));
  unsigned short* wsk = (unsigned short*)(ws + alloc((size_t)4 * 64 * 64 * 2));
  unsigned short* wg1 = (unsigned short*)(ws + alloc((size_t)512 * 32 * 2));
  unsigned short* wg2 = (unsigned short*)(ws + alloc((size_t)256 * 512 * 2));
  unsigned short* wg3 = (unsigned short*)(ws + alloc((size_t)64 * 256 * 2));
  unsigned short* wp1 = (unsigned short*)(ws + alloc((size_t)256 * 64 * 2));
  // overlaid scratch region (lift / spectral / gno never live simultaneously)
  size_t sbase = off;
  float* H1l = (float*)(ws + sbase);                              // NIN*256*4
  size_t so = sbase;
  auto salloc = [&](size_t bytes) -> size_t {
    size_t o = so; so = (so + bytes + 255) & ~(size_t)255; return o;
  };
  float2* T1 = (float2*)(ws + salloc((size_t)32 * 32 * 8 * 64 * 8));
  float2* T2 = (float2*)(ws + salloc((size_t)32 * 16 * 8 * 64 * 8));
  float2* Xf = (float2*)(ws + salloc((size_t)16 * 16 * 8 * 64 * 8));
  float2* Yf = (float2*)(ws + salloc((size_t)16 * 16 * 8 * 64 * 8));
  float2* U1 = (float2*)(ws + salloc((size_t)32 * 16 * 8 * 64 * 8));
  float2* U2 = (float2*)(ws + salloc((size_t)32 * 32 * 8 * 64 * 8));
  float*  sres = (float*)(ws + salloc((size_t)NIN * 64 * 4));
  so = sbase;
  float* H1g = (float*)(ws + salloc((size_t)ECHUNK * 512 * 4));
  float* H2g = (float*)(ws + salloc((size_t)ECHUNK * 256 * 4));
  float* KNg = (float*)(ws + salloc((size_t)ECHUNK * 64 * 4));
  (void)ws_size; (void)in_sizes; (void)n_in; (void)out_size;

  auto gemm = [&](const float* A, const unsigned short* Bt, const float* bias,
                  const float* add, float* Cm, int M, int N, int K, int act) {
    dim3 g(cdiv(N, 64), cdiv(M, 128));
    k_gemm<<<g, 256, 0, stream>>>(A, Bt, bias, add, Cm, M, N, K, act);
  };

  // ---- weight conversion (bf16, [N][Kpad] row-major) ----
  k_wT<<<cdiv(256 * 32, 256),  256, 0, stream>>>(lift_w1, wl1, 256, 6, 32);
  k_wT<<<cdiv(64 * 256, 256),  256, 0, stream>>>(lift_w2, wl2, 64, 256, 256);
  for (int l = 0; l < 4; ++l)
    k_wT<<<cdiv(64 * 64, 256), 256, 0, stream>>>(skip_w + (size_t)l * 64 * 64,
                                                 wsk + (size_t)l * 64 * 64, 64, 64, 64);
  k_wT<<<cdiv(512 * 32, 256),  256, 0, stream>>>(gno_w1, wg1, 512, 6, 32);
  k_wT<<<cdiv(256 * 512, 256), 256, 0, stream>>>(gno_w2, wg2, 256, 512, 512);
  k_wT<<<cdiv(64 * 256, 256),  256, 0, stream>>>(gno_w3, wg3, 64, 256, 256);
  k_wT<<<cdiv(256 * 64, 256),  256, 0, stream>>>(proj_w1, wp1, 256, 64, 64);

  // ---- FNO lifting ----
  k_build_a6<<<cdiv(NIN, 256), 256, 0, stream>>>(f, in_p, A6);
  gemm(A6,  wl1, lift_b1, nullptr, H1l, NIN, 256, 32,  1);
  gemm(H1l, wl2, lift_b2, nullptr, xA,  NIN, 64,  256, 0);

  // ---- FNO layers ----
  float* cur = xA; float* nxt = xB;
  for (int l = 0; l < 4; ++l) {
    k_dft_z  <<<1024, 256, 0, stream>>>(cur, T1);
    k_dft_y  <<< 512, 256, 0, stream>>>(T1, T2);
    k_dft_x  <<< 256, 256, 0, stream>>>(T2, Xf);
    k_modemul<<<2048,  64, 0, stream>>>(Xf,
        spec_wr + (size_t)l * 4 * 64 * 64 * 512,
        spec_wi + (size_t)l * 4 * 64 * 64 * 512, Yf);
    k_idft_x <<< 512, 256, 0, stream>>>(Yf, U1);
    k_idft_y <<<1024, 256, 0, stream>>>(U1, U2);
    k_idft_z <<<1024, 256, 0, stream>>>(U2, sres);
    gemm(cur, wsk + (size_t)l * 64 * 64, skip_b + (size_t)l * 64, sres,
         nxt, NIN, 64, 64, (l < 3) ? 1 : 0);
    float* t = cur; cur = nxt; nxt = t;
  }
  const float* latent = cur;   // == xA after 4 swaps

  // ---- GNO ----
  k_build_feats<<<cdiv(EDGES, 256), 256, 0, stream>>>(in_p, out_p, nbi, seg, feats);
  k_zero<<<cdiv(NOUTP * 64, 256), 256, 0, stream>>>(ssum, NOUTP * 64);
  k_zero<<<cdiv(NOUTP, 256),      256, 0, stream>>>(cnt,  NOUTP);
  k_counts<<<cdiv(EDGES, 256),    256, 0, stream>>>(seg, cnt);
  for (int ci = 0; ci < EDGES / ECHUNK; ++ci) {
    int e0 = ci * ECHUNK;
    gemm(feats + (size_t)e0 * 32, wg1, gno_b1, nullptr, H1g, ECHUNK, 512, 32,  1);
    gemm(H1g,                     wg2, gno_b2, nullptr, H2g, ECHUNK, 256, 512, 1);
    gemm(H2g,                     wg3, gno_b3, nullptr, KNg, ECHUNK, 64,  256, 0);
    k_scatter<<<(ECHUNK * 64) / 256, 256, 0, stream>>>(KNg, latent, nbi, seg, ssum, e0);
  }
  k_mean<<<cdiv(NOUTP * 64, 256), 256, 0, stream>>>(ssum, cnt, outf);

  // ---- projection ----
  gemm(outf, wp1, proj_b1, nullptr, Pj, NOUTP, 256, 64, 1);
  k_proj2<<<(NOUTP * 32) / 256, 256, 0, stream>>>(Pj, proj_w2, proj_b2, out);
}
